// SoftWeightedDecompositionKernel_23914377904551
// MI455X (gfx1250) — matrix operand order repaired and checked by reading, not compiled, not run on hardware
//
#include <hip/hip_runtime.h>
#include <hip/hip_bf16.h>

typedef __attribute__((ext_vector_type(16))) __bf16 v16bf;
typedef __attribute__((ext_vector_type(8)))  __bf16 v8bf;
typedef __attribute__((ext_vector_type(8)))  float  v8f;

#define L_DIM   128
#define N_SYM   30
#define SUB_DIM 16
#define N_ROWS  768

// ---------- helpers ----------
__device__ __forceinline__ unsigned short f2bf(float f) {
    unsigned int b = __float_as_uint(f);
    unsigned int lsb = (b >> 16) & 1u;
    b += 0x7FFFu + lsb;                 // round-to-nearest-even
    return (unsigned short)(b >> 16);
}
__device__ __forceinline__ float bf2f(unsigned short u) {
    return __uint_as_float(((unsigned int)u) << 16);
}

// ---------- workspace layout (byte offsets) ----------
// S   : 900  f32  @ 0
// wf  : 16384 f32 @ 4096
// wb  : 16384 u16 @ 69632
// inv1: 768  f32  @ 102400
// inv2: 768  f32  @ 105472
#define WS_S    0
#define WS_WF   4096
#define WS_WB   69632
#define WS_INV1 102400
#define WS_INV2 105472

// ---------- prep 1: S = A A^T ; w = sigmoid(W) (f32 + bf16) ----------
__global__ __launch_bounds__(256) void swd_prep1(const float* __restrict__ A,
                                                 const float* __restrict__ w_vec,
                                                 float* __restrict__ S,
                                                 float* __restrict__ wf,
                                                 unsigned short* __restrict__ wb) {
    int t = threadIdx.x;
    for (int idx = t; idx < N_SYM * N_SYM; idx += 256) {
        int p = idx / N_SYM, q = idx % N_SYM;
        float s = 0.f;
        #pragma unroll
        for (int d = 0; d < SUB_DIM; ++d) s += A[p * SUB_DIM + d] * A[q * SUB_DIM + d];
        S[idx] = s;
    }
    for (int idx = t; idx < L_DIM * L_DIM; idx += 256) {
        int l = idx >> 7, m = idx & 127;
        float Wv = 0.f;
        if (l != m) {
            int hi = l > m ? l : m;
            int lo = l > m ? m : l;
            Wv = w_vec[hi * (hi - 1) / 2 + lo];
        }
        float sg = 1.f / (1.f + __expf(-Wv));
        wf[idx] = sg;
        wb[idx] = f2bf(sg);
    }
}

// ---------- prep 2: inv_k = rsqrt(d^T w d) for each row of X1 / X2 ----------
__global__ __launch_bounds__(128) void swd_prep2(const int* __restrict__ X1,
                                                 const int* __restrict__ X2,
                                                 const float* __restrict__ S,
                                                 const float* __restrict__ wf,
                                                 float* __restrict__ inv1,
                                                 float* __restrict__ inv2) {
    __shared__ float dsh[L_DIM];
    __shared__ float red[L_DIM];
    int b = blockIdx.x;
    int m = threadIdx.x;
    const int* X = (b < N_ROWS) ? X1 : X2;
    int row = (b < N_ROWS) ? b : b - N_ROWS;
    int xv = X[row * L_DIM + m];
    dsh[m] = S[xv * (N_SYM + 1)];       // S[x, x]
    __syncthreads();
    float s = 0.f;
    for (int l = 0; l < L_DIM; ++l) s += dsh[l] * wf[l * L_DIM + m];
    red[m] = dsh[m] * s;
    __syncthreads();
    for (int off = 64; off > 0; off >>= 1) {
        if (m < off) red[m] += red[m + off];
        __syncthreads();
    }
    if (m == 0) {
        float* o = (b < N_ROWS) ? inv1 : inv2;
        o[row] = rsqrtf(red[0]);
    }
}

// ---------- main: 16x16 output tile per block, quadratic form via WMMA ----------
// dynamic LDS layout (bytes):
//   w_lds : 0      .. 32768   (128*128 bf16)
//   u_lds : 32768  .. 98304   (256 pairs * 128 bf16)
//   S_lds : 98304  .. 101904  (900 f32)
//   x1_lds: 101904 .. 110096  (16*128 int)
//   x2_lds: 110096 .. 118288  (16*128 int)
#define SMEM_BYTES 118288

__global__ __launch_bounds__(256) void swd_main(const int* __restrict__ X1,
                                                const int* __restrict__ X2,
                                                const float* __restrict__ a,
                                                const unsigned short* __restrict__ wb,
                                                const float* __restrict__ S,
                                                const float* __restrict__ inv1,
                                                const float* __restrict__ inv2,
                                                float* __restrict__ out) {
    extern __shared__ char smem[];
    unsigned short* w_lds  = (unsigned short*)(smem);
    unsigned short* u_lds  = (unsigned short*)(smem + 32768);
    float*          S_lds  = (float*)(smem + 98304);
    int*            x1_lds = (int*)(smem + 101904);
    int*            x2_lds = (int*)(smem + 110096);

    const int t  = threadIdx.x;
    const int i0 = blockIdx.y * 16;
    const int j0 = blockIdx.x * 16;

    // ---- stage w (bf16), S, and X tiles ----
    {
        const uint4* src = (const uint4*)wb;
        uint4* dst = (uint4*)w_lds;
        for (int idx = t; idx < (L_DIM * L_DIM) / 8; idx += 256) dst[idx] = src[idx];
    }
    for (int idx = t; idx < N_SYM * N_SYM; idx += 256) S_lds[idx] = S[idx];
    for (int idx = t; idx < 16 * L_DIM; idx += 256) {
        int r = idx >> 7, l = idx & 127;
        x1_lds[idx] = X1[(i0 + r) * L_DIM + l];
        x2_lds[idx] = X2[(j0 + r) * L_DIM + l];
    }
    __syncthreads();

    // ---- gather u[p][l] = S[x1[i,l], x2[j,l]]  (p = i_local*16 + j_local) ----
    {
        int il = t >> 4, jl = t & 15;
        const int* xr1 = &x1_lds[il * L_DIM];
        const int* xr2 = &x2_lds[jl * L_DIM];
        unsigned short* urow = &u_lds[t * L_DIM];
        for (int l = 0; l < L_DIM; ++l) {
            int sidx = xr1[l] * N_SYM + xr2[l];
            urow[l] = f2bf(S_lds[sidx]);
        }
    }
    __syncthreads();

    // ---- quadratic form: K[p] = sum_m (U w)[p,m] * U[p,m] ----
    const int lane = t & 31;
    const int wid  = t >> 5;
    const int hi   = lane >> 4;     // 0: lanes 0-15, 1: lanes 16-31
    const int ln   = lane & 15;
    const float a0 = a[0];
    const float a2 = a0 * a0;

    for (int g = wid; g < 16; g += 8) {          // group g == i_local
        // hoist A operands: rows of U for this group, ISA 16-bit A layout
        v16bf aa[4];
        const __bf16* ubase = (const __bf16*)u_lds + (g * 16 + ln) * L_DIM + hi * 8;
        #pragma unroll
        for (int kk = 0; kk < 4; ++kk) {
            v8bf lo8 = *(const v8bf*)(ubase + kk * 32);
            v8bf hi8 = *(const v8bf*)(ubase + kk * 32 + 16);
            union { v16bf v; v8bf h[2]; } U;
            U.h[0] = lo8; U.h[1] = hi8;
            aa[kk] = U.v;
        }

        float acc[8];
        #pragma unroll
        for (int r = 0; r < 8; ++r) acc[r] = 0.f;

        for (int nb = 0; nb < 8; ++nb) {         // m-block
            v8f c = {};
            #pragma unroll
            for (int kk = 0; kk < 4; ++kk) {     // K = l, 4 chunks of 32
                const __bf16* bp = (const __bf16*)w_lds + (nb * 16 + ln) * L_DIM + kk * 32 + hi * 8;
                union { v16bf v; v8bf h[2]; } BU;
                BU.h[0] = *(const v8bf*)bp; BU.h[1] = *(const v8bf*)(bp + 16);
                v16bf bb = BU.v;
                c = __builtin_amdgcn_wmma_f32_16x16x32_bf16(
                        false, aa[kk], false, bb, (short)0, c, false, false);
            }
            // consume: acc[r] += Y[M=r+8*hi, N=ln] * U[p, m]
            #pragma unroll
            for (int r = 0; r < 8; ++r) {
                int M = r + 8 * hi;              // j_local
                float um = bf2f(u_lds[(g * 16 + M) * L_DIM + nb * 16 + ln]);
                acc[r] += c[r] * um;
            }
        }

        // cross-lane reduction over the 16 N-lanes of each half-wave
        #pragma unroll
        for (int r = 0; r < 8; ++r) {
            float v = acc[r];
            v += __shfl_xor(v, 1);
            v += __shfl_xor(v, 2);
            v += __shfl_xor(v, 4);
            v += __shfl_xor(v, 8);
            if (ln == 0) {
                int jl = r + 8 * hi;
                int gi = i0 + g;
                int gj = j0 + jl;
                out[gi * N_ROWS + gj] = v * inv1[gi] * inv2[gj] * a2;
            }
        }
    }
}

extern "C" void kernel_launch(void* const* d_in, const int* in_sizes, int n_in,
                              void* d_out, int out_size, void* d_ws, size_t ws_size,
                              hipStream_t stream) {
    const int*   X1    = (const int*)d_in[0];
    const int*   X2    = (const int*)d_in[1];
    const float* a     = (const float*)d_in[2];
    const float* A     = (const float*)d_in[3];
    const float* w_vec = (const float*)d_in[4];
    float* out = (float*)d_out;

    char* ws = (char*)d_ws;
    float*          ws_S    = (float*)(ws + WS_S);
    float*          ws_wf   = (float*)(ws + WS_WF);
    unsigned short* ws_wb   = (unsigned short*)(ws + WS_WB);
    float*          ws_inv1 = (float*)(ws + WS_INV1);
    float*          ws_inv2 = (float*)(ws + WS_INV2);

    swd_prep1<<<1, 256, 0, stream>>>(A, w_vec, ws_S, ws_wf, ws_wb);
    swd_prep2<<<2 * N_ROWS, 128, 0, stream>>>(X1, X2, ws_S, ws_wf, ws_inv1, ws_inv2);

    dim3 grid(N_ROWS / 16, N_ROWS / 16);
    swd_main<<<grid, 256, SMEM_BYTES, stream>>>(X1, X2, a, ws_wb, ws_S,
                                                ws_inv1, ws_inv2, out);
}
